// AnyAttention_30983894073936
// MI455X (gfx1250) — matrix operand order repaired and checked by reading, not compiled, not run on hardware
//
#include <hip/hip_runtime.h>
#include <hip/hip_bf16.h>

typedef __bf16 bf16;
typedef __attribute__((ext_vector_type(16))) __bf16 bf16x16;
typedef __attribute__((ext_vector_type(8)))  __bf16 bf16x8;
typedef __attribute__((ext_vector_type(8)))  float  f32x8;

#define WMMA_BF16(A, B, Cacc) \
  __builtin_amdgcn_wmma_f32_16x16x32_bf16(false, (A), false, (B), (short)0, (Cacc), false, false)

constexpr int   kC     = 1024;   // model dim
constexpr int   kG     = 16;     // heads
constexpr int   kHd    = 64;     // head dim
constexpr float kEps   = 1e-5f;
constexpr float kScale = 0.125f; // 64^-0.5

// ---------------------------------------------------------------------------
// Fused pos-add + LayerNorm, fp32 in -> bf16 out. One 256-thread block per row.
// ---------------------------------------------------------------------------
__global__ void ln_bf16_kernel(const float* __restrict__ x,
                               const float* __restrict__ pos,
                               const float* __restrict__ w,
                               const float* __restrict__ bptr,
                               bf16* __restrict__ out) {
  const int    row  = blockIdx.x;
  const int    tid  = threadIdx.x;
  const size_t base = (size_t)row * kC;

  float v[4];
  float s = 0.f, ss = 0.f;
#pragma unroll
  for (int i = 0; i < 4; ++i) {
    int   c = tid + i * 256;
    float t = x[base + c];
    if (pos) t += pos[base + c];
    v[i] = t;
    s += t;
    ss += t * t;
  }
  // wave32 reduction
#pragma unroll
  for (int off = 1; off < 32; off <<= 1) {
    s  += __shfl_xor(s, off, 32);
    ss += __shfl_xor(ss, off, 32);
  }
  __shared__ float rs[8], rss[8];
  const int wid = tid >> 5;
  if ((tid & 31) == 0) { rs[wid] = s; rss[wid] = ss; }
  __syncthreads();
  float ts = 0.f, tss = 0.f;
#pragma unroll
  for (int i = 0; i < 8; ++i) { ts += rs[i]; tss += rss[i]; }

  const float mu   = ts / (float)kC;
  const float var  = tss / (float)kC - mu * mu;
  const float rstd = rsqrtf(var + kEps);
#pragma unroll
  for (int i = 0; i < 4; ++i) {
    int c = tid + i * 256;
    out[base + c] = (bf16)((v[i] - mu) * rstd * w[c] + bptr[c]);
  }
}

// ---------------------------------------------------------------------------
// fp32 -> bf16 weight conversion
// ---------------------------------------------------------------------------
__global__ void f32_to_bf16_kernel(const float* __restrict__ in,
                                   bf16* __restrict__ out, int n) {
  int i = blockIdx.x * 256 + threadIdx.x;
  if (i < n) out[i] = (bf16)in[i];
}

// ---------------------------------------------------------------------------
// Per-head transpose: v4 [B*Lk, C] -> v4t [B, G, Hd, Lk], so that in the
// attention P@V step every B-fragment column (fixed head-dim c, varying k)
// is a contiguous 32-byte run. LDS-tiled: both global read & write coalesced.
// One block per (b, h, 64-wide k tile).
// ---------------------------------------------------------------------------
__global__ void transpose_v_kernel(const bf16* __restrict__ in,
                                   bf16* __restrict__ out, int Lk) {
  __shared__ bf16 tile[64][72];  // +8 pad to spread LDS banks
  const int nk = Lk / 64;
  const int kt = blockIdx.x % nk;
  const int h  = (blockIdx.x / nk) % kG;
  const int b  = blockIdx.x / (nk * kG);
  const int k0 = kt * 64;
  const int tid = threadIdx.x;

#pragma unroll
  for (int i = 0; i < 16; ++i) {
    int idx = i * 256 + tid;
    int r   = idx >> 6;   // k within tile
    int c   = idx & 63;   // head-dim
    tile[r][c] = in[(size_t)(b * Lk + k0 + r) * kC + h * kHd + c];
  }
  __syncthreads();
#pragma unroll
  for (int i = 0; i < 16; ++i) {
    int idx = i * 256 + tid;
    int c   = idx >> 6;   // head-dim (output row)
    int kk  = idx & 63;   // k (output col)
    out[((size_t)(b * kG + h) * kHd + c) * Lk + k0 + kk] = tile[kk][c];
  }
}

// ---------------------------------------------------------------------------
// Y[m,d] = sum_c X[m,c] * W[d,c]  (torch Linear, W row-major [out,in])
// bf16 inputs, f32 WMMA accumulate. Each wave computes a 16x64 strip:
// one A fragment reused across 4 B tiles per K step of 32.
// A layout (16-bit A 16x32, as loaded here): lane%16 = row M; lane/16 selects
// K 0..15 vs 16..31, elems 0..15 are 16 contiguous K values (same order as B).
// B layout (32x16): lane%16 = column N; lane/16 selects K 0..15 vs 16..31,
// elems 0..15 are 16 contiguous K values. Column N of B == row N of W (contig).
// ---------------------------------------------------------------------------
__global__ void gemm_bf16_wmma_kernel(const bf16* __restrict__ X,
                                      const bf16* __restrict__ W,
                                      bf16* __restrict__ Ybf,
                                      float* __restrict__ Yf,
                                      const float* __restrict__ bias,
                                      int M, int D) {
  const int wid  = (blockIdx.x * blockDim.x + threadIdx.x) >> 5;
  const int lane = threadIdx.x & 31;
  const int nG   = D / 64;
  const int tm   = wid / nG;
  const int ng   = wid % nG;
  if (tm * 16 >= M) return;

  const int half = lane >> 4;
  const int l16  = lane & 15;

  const bf16* arow = X + (size_t)(tm * 16 + l16) * kC;
  const bf16* brow[4];
#pragma unroll
  for (int t = 0; t < 4; ++t)
    brow[t] = W + (size_t)(ng * 64 + t * 16 + l16) * kC;

  f32x8 acc[4] = {};
  for (int k0 = 0; k0 < kC; k0 += 32) {
    bf16x16 a;
    a = *(const bf16x16*)(arow + k0 + half * 16);
#pragma unroll
    for (int t = 0; t < 4; ++t) {
      bf16x16 bm = *(const bf16x16*)(brow[t] + k0 + half * 16);
      acc[t] = WMMA_BF16(a, bm, acc[t]);
    }
  }

  // C/D layout: VGPR v, row M = v + 8*(lane/16), col N = lane%16
  if (Yf) {
#pragma unroll
    for (int t = 0; t < 4; ++t)
#pragma unroll
      for (int v = 0; v < 8; ++v) {
        int r = tm * 16 + v + half * 8;
        int c = ng * 64 + t * 16 + l16;
        Yf[(size_t)r * D + c] = acc[t][v] + (bias ? bias[c] : 0.f);
      }
  } else {
#pragma unroll
    for (int t = 0; t < 4; ++t)
#pragma unroll
      for (int v = 0; v < 8; ++v) {
        int r = tm * 16 + v + half * 8;
        int c = ng * 64 + t * 16 + l16;
        Ybf[(size_t)r * D + c] = (bf16)acc[t][v];
      }
  }
}

// ---------------------------------------------------------------------------
// Streaming (flash-style) attention. One wave per (batch, head, 16-row q tile).
// attn[b,q,h,k] = <q4[b,q,h,:], k4[b,k,h,:]> * SCALE, masked -> -inf;
// online softmax over k; O = P @ Vt, normalized at the end.
// V comes pre-transposed (v4t [B,G,Hd,Lk]) so B-fragments are contiguous.
// ---------------------------------------------------------------------------
__global__ void attn_wmma_kernel(const bf16* __restrict__ q4,
                                 const bf16* __restrict__ k4,
                                 const bf16* __restrict__ v4t,
                                 const int* __restrict__ mask,
                                 bf16* __restrict__ out,
                                 int Lq, int Lk) {
  __shared__ __align__(32) bf16 pbuf[8 * 16 * 32];  // per-wave 16x32 P staging

  const int wslot = threadIdx.x >> 5;
  const int lane  = threadIdx.x & 31;
  const int gw    = blockIdx.x * 8 + wslot;
  const int qt    = gw % (Lq / 16);
  const int h     = (gw / (Lq / 16)) % kG;
  const int b     = gw / ((Lq / 16) * kG);
  const int half  = lane >> 4;
  const int l16   = lane & 15;
  bf16* pw = pbuf + wslot * 512;

  // Q fragment: rows qt*16..+15, head dim 64 -> two 16x32 A fragments
  const bf16* qrow = q4 + (size_t)(b * Lq + qt * 16 + l16) * kC + h * kHd;
  bf16x16 aq0, aq1;
  aq0 = *(const bf16x16*)(qrow + half * 16);
  aq1 = *(const bf16x16*)(qrow + 32 + half * 16);

  // Per-lane V row base: column n = l16 selects head-dim c = t*16 + l16,
  // row in v4t is ((b*G + h)*Hd + c), contiguous along k.
  const bf16* vbase = v4t + ((size_t)(b * kG + h) * kHd + l16) * Lk;

  f32x8 o[4] = {};
  float mrow[8], lrow[8];
#pragma unroll
  for (int v = 0; v < 8; ++v) { mrow[v] = -1e30f; lrow[v] = 0.f; }

  for (int kb = 0; kb < Lk; kb += 32) {
    // scores for the two 16-wide k subtiles
    f32x8 s[2];
#pragma unroll
    for (int j = 0; j < 2; ++j) {
      const bf16* krow =
          k4 + (size_t)(b * Lk + kb + j * 16 + l16) * kC + h * kHd;
      bf16x16 bk0 = *(const bf16x16*)(krow + half * 16);       // c 0..31
      bf16x16 bk1 = *(const bf16x16*)(krow + 32 + half * 16);  // c 32..63
      f32x8 z = {};
      z = WMMA_BF16(aq0, bk0, z);
      z = WMMA_BF16(aq1, bk1, z);
      s[j] = z;
    }
    // scale + mask (mask column == this lane's N = l16)
#pragma unroll
    for (int j = 0; j < 2; ++j) {
      int col = kb + j * 16 + l16;
      int mv  = mask[(size_t)b * Lk + col];
#pragma unroll
      for (int v = 0; v < 8; ++v)
        s[j][v] = mv ? -1e30f : s[j][v] * kScale;
    }
    // online softmax: row r = v + 8*half lives in one 16-lane half,
    // so xor-shuffles with masks 1,2,4,8 reduce exactly over that row.
    float alpha[8];
#pragma unroll
    for (int v = 0; v < 8; ++v) {
      float cur = fmaxf(s[0][v], s[1][v]);
#pragma unroll
      for (int off = 1; off < 16; off <<= 1)
        cur = fmaxf(cur, __shfl_xor(cur, off, 32));
      float mn = fmaxf(mrow[v], cur);
      alpha[v] = __expf(mrow[v] - mn);
      float p0 = __expf(s[0][v] - mn);
      float p1 = __expf(s[1][v] - mn);
      s[0][v] = p0;
      s[1][v] = p1;
      float r = p0 + p1;
#pragma unroll
      for (int off = 1; off < 16; off <<= 1)
        r += __shfl_xor(r, off, 32);
      lrow[v] = lrow[v] * alpha[v] + r;
      mrow[v] = mn;
    }
    // rescale accumulators
#pragma unroll
    for (int t = 0; t < 4; ++t)
#pragma unroll
      for (int v = 0; v < 8; ++v) o[t][v] *= alpha[v];

    // Stage P (D-layout) to LDS row-major [16][32], reload in A layout.
#pragma unroll
    for (int v = 0; v < 8; ++v) {
      int r = v + half * 8;
      pw[r * 32 + l16]      = (bf16)s[0][v];
      pw[r * 32 + 16 + l16] = (bf16)s[1][v];
    }
    asm volatile("s_wait_dscnt 0" ::: "memory");  // order LDS store -> load
    bf16x16 ap;
    ap = *(const bf16x16*)(pw + l16 * 32 + half * 16);

    // O += P (16x32) x V (32x64), 4 N tiles; contiguous 32-B B-fragment loads.
#pragma unroll
    for (int t = 0; t < 4; ++t) {
      bf16x16 bv =
          *(const bf16x16*)(vbase + (size_t)t * 16 * Lk + kb + half * 16);
      o[t] = WMMA_BF16(ap, bv, o[t]);
    }
  }

  // normalize + store (all-masked rows get p = 1 everywhere -> l = Lk, output = mean of V)
#pragma unroll
  for (int v = 0; v < 8; ++v) {
    float inv = lrow[v] > 0.f ? 1.f / lrow[v] : 0.f;
    int   r   = b * Lq + qt * 16 + v + half * 8;
#pragma unroll
    for (int t = 0; t < 4; ++t)
      out[(size_t)r * kC + h * kHd + t * 16 + l16] = (bf16)(o[t][v] * inv);
  }
}

// ---------------------------------------------------------------------------
extern "C" void kernel_launch(void* const* d_in, const int* in_sizes, int n_in,
                              void* d_out, int out_size, void* d_ws,
                              size_t ws_size, hipStream_t stream) {
  (void)in_sizes; (void)n_in; (void)out_size; (void)ws_size;

  const int B  = 2;
  const int Lq = 1024;
  const int Lk = 2048;
  const int Mq = B * Lq;   // 2048
  const int Mk = B * Lk;   // 4096

  const float* q      = (const float*)d_in[0];
  const float* k      = (const float*)d_in[1];
  const float* v      = (const float*)d_in[2];
  const float* qpos   = (const float*)d_in[3];
  const float* kpos   = (const float*)d_in[4];
  const int*   mask   = (const int*)d_in[5];
  const float* ln_q_w = (const float*)d_in[6];
  const float* ln_q_b = (const float*)d_in[7];
  const float* ln_k_w = (const float*)d_in[8];
  const float* ln_k_b = (const float*)d_in[9];
  const float* ln_v_w = (const float*)d_in[10];
  const float* ln_v_b = (const float*)d_in[11];
  const float* w_q    = (const float*)d_in[12];
  const float* w_k    = (const float*)d_in[13];
  const float* w_v    = (const float*)d_in[14];
  const float* w_proj = (const float*)d_in[15];
  const float* b_proj = (const float*)d_in[16];
  float*       outp   = (float*)d_out;

  // carve workspace (256-B aligned slots)
  char*  ws  = (char*)d_ws;
  size_t off = 0;
  auto take = [&](size_t bytes) -> void* {
    void* p = ws + off;
    off += (bytes + 255) & ~(size_t)255;
    return p;
  };
  bf16* qn  = (bf16*)take((size_t)Mq * kC * 2);
  bf16* kn  = (bf16*)take((size_t)Mk * kC * 2);
  bf16* vn  = (bf16*)take((size_t)Mk * kC * 2);
  bf16* q4b = (bf16*)take((size_t)Mq * kC * 2);
  bf16* k4b = (bf16*)take((size_t)Mk * kC * 2);
  bf16* v4b = (bf16*)take((size_t)Mk * kC * 2);
  bf16* v4t = (bf16*)take((size_t)Mk * kC * 2);  // per-head transposed V
  bf16* att = (bf16*)take((size_t)Mq * kC * 2);
  bf16* wqb = (bf16*)take((size_t)kC * kC * 2);
  bf16* wkb = (bf16*)take((size_t)kC * kC * 2);
  bf16* wvb = (bf16*)take((size_t)kC * kC * 2);
  bf16* wpb = (bf16*)take((size_t)kC * kC * 2);

  // 1) weight conversion
  const int nw = kC * kC;
  f32_to_bf16_kernel<<<nw / 256, 256, 0, stream>>>(w_q, wqb, nw);
  f32_to_bf16_kernel<<<nw / 256, 256, 0, stream>>>(w_k, wkb, nw);
  f32_to_bf16_kernel<<<nw / 256, 256, 0, stream>>>(w_v, wvb, nw);
  f32_to_bf16_kernel<<<nw / 256, 256, 0, stream>>>(w_proj, wpb, nw);

  // 2) pos-add + LayerNorm (qpos/kpos flatten to the same [row, C] layout)
  ln_bf16_kernel<<<Mq, 256, 0, stream>>>(q, qpos, ln_q_w, ln_q_b, qn);
  ln_bf16_kernel<<<Mk, 256, 0, stream>>>(k, kpos, ln_k_w, ln_k_b, kn);
  ln_bf16_kernel<<<Mk, 256, 0, stream>>>(v, nullptr, ln_v_w, ln_v_b, vn);

  // 3) projections: waves = (M/16)*(C/64) = M ; 8 waves/block
  gemm_bf16_wmma_kernel<<<Mq / 8, 256, 0, stream>>>(qn, wqb, q4b, nullptr,
                                                    nullptr, Mq, kC);
  gemm_bf16_wmma_kernel<<<Mk / 8, 256, 0, stream>>>(kn, wkb, k4b, nullptr,
                                                    nullptr, Mk, kC);
  gemm_bf16_wmma_kernel<<<Mk / 8, 256, 0, stream>>>(vn, wvb, v4b, nullptr,
                                                    nullptr, Mk, kC);

  // 3b) per-head transpose of V for contiguous P@V fragments
  transpose_v_kernel<<<B * kG * (Lk / 64), 256, 0, stream>>>(v4b, v4t, Lk);

  // 4) attention: waves = B * heads * (Lq/16) = 2048 ; 8 waves/block
  attn_wmma_kernel<<<(B * kG * (Lq / 16)) / 8, 256, 0, stream>>>(
      q4b, k4b, v4t, mask, att, Lq, Lk);

  // 5) output projection + bias -> f32 d_out
  gemm_bf16_wmma_kernel<<<Mq / 8, 256, 0, stream>>>(att, wpb, nullptr, outp,
                                                    b_proj, Mq, kC);
}
